// TwinPolicy_2199023255725
// MI455X (gfx1250) — matrix operand order repaired and checked by reading, not compiled, not run on hardware
//
#include <hip/hip_runtime.h>

typedef __attribute__((ext_vector_type(16))) _Float16 v16h;
typedef __attribute__((ext_vector_type(8)))  _Float16 v8h;
typedef __attribute__((ext_vector_type(8)))  float    v8f;

#define NODES   4096
#define DHID    256
#define INDIM   1024

// ---------------------------------------------------------------------------
// Fragment loaders (pure b128 loads, no VALU):
//   A 16x32 f16 : lane<16 row=lane holds K{0..7,16..23}; lane>=16 K{8..15,24..31}
//     -> two contiguous 8xf16 (16B) loads at +half*8 and +16+half*8
//   B 32x16 f16 : lane<16 col=lane holds K{0..7,16..23}; lane>=16 K{8..15,24..31}
//     -> two contiguous 8xf16 (16B) loads at +half*8 and +16+half*8
// ---------------------------------------------------------------------------
__device__ __forceinline__ v16h load_afrag(const _Float16* __restrict__ p, int half) {
  v8h c0 = *(const v8h*)(p + half * 8);
  v8h c1 = *(const v8h*)(p + 16 + half * 8);
  return __builtin_shufflevector(c0, c1, 0, 1, 2, 3, 4, 5, 6, 7,
                                 8, 9, 10, 11, 12, 13, 14, 15);
}
__device__ __forceinline__ v16h load_bfrag(const _Float16* __restrict__ p, int half) {
  v8h c0 = *(const v8h*)(p + half * 8);
  v8h c1 = *(const v8h*)(p + 16 + half * 8);
  return __builtin_shufflevector(c0, c1, 0, 1, 2, 3, 4, 5, 6, 7,
                                 8, 9, 10, 11, 12, 13, 14, 15);
}

// ---------------------------------------------------------------------------
// f32 -> f16 pre-conversion (bandwidth-trivial, runs once per operand).
// ---------------------------------------------------------------------------
__global__ __launch_bounds__(256)
void cvt_f16_k(const float* __restrict__ in, _Float16* __restrict__ out, int n4) {
  const int idx = blockIdx.x * blockDim.x + threadIdx.x;
  if (idx >= n4) return;
  const float4 v = ((const float4*)in)[idx];
  v8h o;  // write 4 halves via 8B store
  out[4 * idx + 0] = (_Float16)v.x;
  out[4 * idx + 1] = (_Float16)v.y;
  out[4 * idx + 2] = (_Float16)v.z;
  out[4 * idx + 3] = (_Float16)v.w;
  (void)o;
}

// ---------------------------------------------------------------------------
// WMMA GEMM, all-f16 operands: out[M,N] f16 = act(A[M,K] * W[N,K]^T + bias).
// One wave computes a 16x64 output strip: 4 independent accumulator chains
// (NACC=4) so consecutive v_wmma never hit the same D->C chain, and the A
// fragment is reused 4x per K-step.
// ---------------------------------------------------------------------------
template <bool RELU>
__global__ __launch_bounds__(256)
void gemm_f16(const _Float16* __restrict__ A, const _Float16* __restrict__ W,
              const float* __restrict__ bias, _Float16* __restrict__ out,
              int M, int K, int Nn) {
  const int wave   = blockIdx.x * (blockDim.x >> 5) + (threadIdx.x >> 5);
  const int ngroup = Nn >> 6;                     // groups of 4 N-tiles
  const int tm  = wave / ngroup;
  const int tng = wave % ngroup;
  if (tm * 16 >= M) return;                       // wave-uniform guard
  const int lane = threadIdx.x & 31;
  const int half = lane >> 4;
  const int r    = lane & 15;

  const _Float16* arow = A + (size_t)(tm * 16 + r) * K;
  const _Float16* wcol[4];
#pragma unroll
  for (int c = 0; c < 4; ++c)
    wcol[c] = W + (size_t)(tng * 64 + c * 16 + r) * K;

  v8f acc[4] = {v8f{}, v8f{}, v8f{}, v8f{}};
#pragma unroll 4
  for (int kb = 0; kb < K; kb += 32) {
    const v16h a = load_afrag(arow + kb, half);
#pragma unroll
    for (int c = 0; c < 4; ++c) {
      const v16h b = load_bfrag(wcol[c] + kb, half);
      acc[c] = __builtin_amdgcn_wmma_f32_16x16x32_f16(false, a, false, b,
                                                      (short)0, acc[c], false, false);
    }
  }

#pragma unroll
  for (int c = 0; c < 4; ++c) {
    const int col = tng * 64 + c * 16 + r;
    const float bv = bias[col];
#pragma unroll
    for (int g = 0; g < 8; ++g) {
      const int row = tm * 16 + g + 8 * half;
      float v = acc[c][g] + bv;
      if (RELU) v = fmaxf(v, 0.0f);
      out[(size_t)row * Nn + col] = (_Float16)v;
    }
  }
}

// ---------------------------------------------------------------------------
// Edge MLP: one wave per 16 edges.  A rows are gathered [h_i | h_j] (K=512,
// f16).  4 outer passes of 4 accumulator chains cover N=256; ReLU + be1 + dot
// with We2 fused into each pass epilogue.  K loop split (h_i half / h_j half)
// to avoid per-iteration selects.
// ---------------------------------------------------------------------------
__global__ __launch_bounds__(256)
void edge_score(const _Float16* __restrict__ h, const int* __restrict__ ei,
                const _Float16* __restrict__ We1h, const float* __restrict__ be1,
                const float* __restrict__ We2, const float* __restrict__ be2,
                float* __restrict__ s, int E_) {
  const int wave = blockIdx.x * (blockDim.x >> 5) + (threadIdx.x >> 5);
  if (wave * 16 >= E_) return;                    // wave-uniform guard
  const int lane = threadIdx.x & 31;
  const int half = lane >> 4;
  const int r    = lane & 15;

  const int et = wave * 16 + r;                   // edge row handled by lane
  const _Float16* hi = h + (size_t)ei[2 * et]     * DHID;
  const _Float16* hj = h + (size_t)ei[2 * et + 1] * DHID;

  float part[8] = {0, 0, 0, 0, 0, 0, 0, 0};

  for (int ng = 0; ng < 4; ++ng) {
    const _Float16* wcol[4];
#pragma unroll
    for (int c = 0; c < 4; ++c)
      wcol[c] = We1h + (size_t)(ng * 64 + c * 16 + r) * 512;

    v8f acc[4] = {v8f{}, v8f{}, v8f{}, v8f{}};
#pragma unroll
    for (int kb = 0; kb < 8; ++kb) {              // K 0..255 from h_i
      const v16h a = load_afrag(hi + kb * 32, half);
#pragma unroll
      for (int c = 0; c < 4; ++c) {
        const v16h b = load_bfrag(wcol[c] + kb * 32, half);
        acc[c] = __builtin_amdgcn_wmma_f32_16x16x32_f16(false, a, false, b,
                                                        (short)0, acc[c], false, false);
      }
    }
#pragma unroll
    for (int kb = 0; kb < 8; ++kb) {              // K 256..511 from h_j
      const v16h a = load_afrag(hj + kb * 32, half);
#pragma unroll
      for (int c = 0; c < 4; ++c) {
        const v16h b = load_bfrag(wcol[c] + (kb + 8) * 32, half);
        acc[c] = __builtin_amdgcn_wmma_f32_16x16x32_f16(false, a, false, b,
                                                        (short)0, acc[c], false, false);
      }
    }

#pragma unroll
    for (int c = 0; c < 4; ++c) {
      const int n = ng * 64 + c * 16 + r;
      const float bn = be1[n];
      const float w2 = We2[n];
#pragma unroll
      for (int g = 0; g < 8; ++g)
        part[g] += fmaxf(acc[c][g] + bn, 0.0f) * w2;
    }
  }

  // Reduce the 16 lanes within each half (each half holds 8 edge rows).
#pragma unroll
  for (int g = 0; g < 8; ++g) {
    float v = part[g];
    v += __shfl_xor(v, 1, 32);
    v += __shfl_xor(v, 2, 32);
    v += __shfl_xor(v, 4, 32);
    v += __shfl_xor(v, 8, 32);
    part[g] = v;
  }
  if (r < 8) {
    float v = part[0];
#pragma unroll
    for (int g = 1; g < 8; ++g)
      if (r == g) v = part[g];
    s[wave * 16 + r + 8 * half] = v + be2[0];
  }
}

// ---------------------------------------------------------------------------
// Softmax reductions (single block each), zero-fill, and scatter.
// ---------------------------------------------------------------------------
__global__ __launch_bounds__(1024)
void reduce_max_k(const float* __restrict__ s, int n, float* __restrict__ scal) {
  __shared__ float red[1024];
  float m = -3.402823466e38f;
  for (int idx = threadIdx.x; idx < n; idx += blockDim.x) m = fmaxf(m, s[idx]);
  red[threadIdx.x] = m;
  __syncthreads();
  for (int off = blockDim.x >> 1; off > 0; off >>= 1) {
    if (threadIdx.x < off)
      red[threadIdx.x] = fmaxf(red[threadIdx.x], red[threadIdx.x + off]);
    __syncthreads();
  }
  if (threadIdx.x == 0) scal[0] = red[0];
}

__global__ __launch_bounds__(1024)
void reduce_sumexp_k(const float* __restrict__ s, int n, float* __restrict__ scal) {
  __shared__ float red[1024];
  const float m = scal[0];
  float acc = 0.0f;
  for (int idx = threadIdx.x; idx < n; idx += blockDim.x)
    acc += __expf(s[idx] - m);
  red[threadIdx.x] = acc;
  __syncthreads();
  for (int off = blockDim.x >> 1; off > 0; off >>= 1) {
    if (threadIdx.x < off)
      red[threadIdx.x] += red[threadIdx.x + off];
    __syncthreads();
  }
  if (threadIdx.x == 0) scal[1] = red[0];
}

__global__ __launch_bounds__(256)
void zero_out_k(float4* __restrict__ out, int n4) {
  const int idx = blockIdx.x * blockDim.x + threadIdx.x;
  if (idx < n4) out[idx] = make_float4(0.f, 0.f, 0.f, 0.f);
}

__global__ __launch_bounds__(256)
void scatter_k(const float* __restrict__ s, const int* __restrict__ ei,
               const float* __restrict__ scal, float* __restrict__ out, int n) {
  const int e = blockIdx.x * blockDim.x + threadIdx.x;
  if (e >= n) return;
  const float p = __expf(s[e] - scal[0]) / scal[1];
  out[(size_t)ei[2 * e] * NODES + ei[2 * e + 1]] = p;
}

// ---------------------------------------------------------------------------
extern "C" void kernel_launch(void* const* d_in, const int* in_sizes, int n_in,
                              void* d_out, int out_size, void* d_ws, size_t ws_size,
                              hipStream_t stream) {
  const float* ideal = (const float*)d_in[0];
  const int*   ei    = (const int*)d_in[1];
  const float* W1    = (const float*)d_in[2];
  const float* b1    = (const float*)d_in[3];
  const float* W2    = (const float*)d_in[4];
  const float* b2    = (const float*)d_in[5];
  const float* We1   = (const float*)d_in[6];
  const float* be1   = (const float*)d_in[7];
  const float* We2   = (const float*)d_in[8];
  const float* be2   = (const float*)d_in[9];
  float* out = (float*)d_out;

  const int E = in_sizes[1] / 2;

  // Workspace layout (all 256B-aligned)
  char* ws = (char*)d_ws;
  size_t off = 0;
  _Float16* x16  = (_Float16*)(ws + off); off += (size_t)NODES * INDIM * 2;  // 8 MB
  _Float16* w1h  = (_Float16*)(ws + off); off += (size_t)DHID * INDIM * 2;   // 512 KB
  _Float16* w2h  = (_Float16*)(ws + off); off += (size_t)DHID * DHID * 2;    // 128 KB
  _Float16* we1h = (_Float16*)(ws + off); off += (size_t)DHID * 512 * 2;     // 256 KB
  _Float16* h1   = (_Float16*)(ws + off); off += (size_t)NODES * DHID * 2;   // 2 MB
  _Float16* h    = (_Float16*)(ws + off); off += (size_t)NODES * DHID * 2;   // 2 MB
  float*    s    = (float*)(ws + off);    off += (size_t)E * 4;              // 512 KB
  float*    scal = (float*)(ws + off);

  // Zero-fill the 64MB output (independent of the GEMM chain).
  {
    const int n4 = out_size / 4;
    zero_out_k<<<(n4 + 255) / 256, 256, 0, stream>>>((float4*)out, n4);
  }

  // Pre-convert all GEMM operands to f16 (one pass each, HBM/L2-bound).
  cvt_f16_k<<<(NODES * INDIM / 4 + 255) / 256, 256, 0, stream>>>(ideal, x16, NODES * INDIM / 4);
  cvt_f16_k<<<(DHID * INDIM / 4 + 255) / 256, 256, 0, stream>>>(W1, w1h, DHID * INDIM / 4);
  cvt_f16_k<<<(DHID * DHID / 4 + 255) / 256, 256, 0, stream>>>(W2, w2h, DHID * DHID / 4);
  cvt_f16_k<<<(DHID * 512 / 4 + 255) / 256, 256, 0, stream>>>(We1, we1h, DHID * 512 / 4);

  // GEMM1: h1 = ReLU(x16 * W1^T + b1)   (M=4096, K=1024, N=256)
  {
    const int waves = (NODES / 16) * (DHID / 64);  // 1024
    gemm_f16<true><<<waves / 8, 256, 0, stream>>>(x16, w1h, b1, h1, NODES, INDIM, DHID);
  }
  // GEMM2: h = h1 * W2^T + b2            (M=4096, K=256, N=256)
  {
    const int waves = (NODES / 16) * (DHID / 64);  // 1024
    gemm_f16<false><<<waves / 8, 256, 0, stream>>>(h1, w2h, b2, h, NODES, DHID, DHID);
  }
  // Edge MLP: s[e] = ReLU([h_i|h_j] We1^T + be1) . We2 + be2
  {
    const int waves = E / 16;                      // 8192
    edge_score<<<waves / 8, 256, 0, stream>>>(h, ei, we1h, be1, We2, be2, s, E);
  }
  // Softmax over the E finite scores
  reduce_max_k<<<1, 1024, 0, stream>>>(s, E, scal);
  reduce_sumexp_k<<<1, 1024, 0, stream>>>(s, E, scal);
  scatter_k<<<(E + 255) / 256, 256, 0, stream>>>(s, ei, scal, out, E);
}
